// LossStyle_41729902248606
// MI455X (gfx1250) — matrix operand order repaired and checked by reading, not compiled, not run on hardware
//
#include <hip/hip_runtime.h>
#include <hip/hip_bf16.h>
#include <math.h>

typedef __attribute__((ext_vector_type(16))) _Float16 v16h;
typedef __attribute__((ext_vector_type(8)))  _Float16 v8h;
typedef __attribute__((ext_vector_type(8)))  float    v8f;
typedef __attribute__((ext_vector_type(4)))  int      v4i;

#if defined(__has_builtin)
#if __has_builtin(__builtin_amdgcn_global_load_async_to_lds_b128)
#define HAVE_ASYNC_LDS 1
#endif
#endif

typedef __attribute__((address_space(1))) v4i* as1_v4i_p;
typedef __attribute__((address_space(3))) v4i* as3_v4i_p;

__device__ __forceinline__ v16h join16(v8h lo, v8h hi) {
    return __builtin_shufflevector(lo, hi, 0, 1, 2, 3, 4, 5, 6, 7,
                                           8, 9, 10, 11, 12, 13, 14, 15);
}

// ---------------------------------------------------------------------------
// Input normalization: (x - mean_c) / std_c, f32 -> f16
// ---------------------------------------------------------------------------
__global__ void norm_input_kernel(const float* __restrict__ in,
                                  _Float16* __restrict__ out, int HW) {
    int idx = blockIdx.x * blockDim.x + threadIdx.x;
    if (idx >= 3 * HW) return;
    int c = idx / HW;
    const float mean[3] = {0.485f, 0.456f, 0.406f};
    const float stdv[3] = {0.229f, 0.224f, 0.225f};
    out[idx] = (_Float16)((in[idx] - mean[c]) / stdv[c]);
}

// ---------------------------------------------------------------------------
// Weight pre-convert: f32 [Cout][K] -> f16 [Cout][Kp], zero-padded K->Kp
// ---------------------------------------------------------------------------
__global__ void cvt_weights_kernel(const float* __restrict__ w,
                                   _Float16* __restrict__ wh,
                                   int K, int Kp, int total) {
    int e = blockIdx.x * blockDim.x + threadIdx.x;
    if (e >= total) return;
    int row = e / Kp;
    int k = e - row * Kp;
    wh[e] = (k < K) ? (_Float16)w[row * K + k] : (_Float16)0.0f;
}

// ---------------------------------------------------------------------------
// 3x3 SAME conv + bias + ReLU, implicit GEMM on v_wmma_f32_16x16x32_f16.
// Block = 128 threads (4 waves). Wave: 16 Cout rows x 32 pixels (2 WMMA/step).
// grid.x = HW/32 pixel tiles, grid.y = Cout/64. Weights pre-converted f16.
// ---------------------------------------------------------------------------
template <typename InT, typename OutT>
__global__ void conv3x3_wmma(const InT* __restrict__ in,
                             const _Float16* __restrict__ wh,
                             const float* __restrict__ bias,
                             OutT* __restrict__ out,
                             int Cin, int Cout, int H, int W, int Kp) {
    __shared__ alignas(32) _Float16 ldsB[32 * 32];   // [pixel][k]
    const int tid  = threadIdx.x;
    const int wave = tid >> 5;
    const int lane = tid & 31;
    const int hi   = lane >> 4;
    const int nloc = lane & 15;
    const int HW   = H * W;
    const int j0   = blockIdx.x * 32;               // pixel tile base
    const int mBase = blockIdx.y * 64 + wave * 16;  // Cout tile base
    const int K    = Cin * 9;
    const int row  = mBase + nloc;

    v8f acc0 = {}, acc1 = {};
    for (int k0 = 0; k0 < Kp; k0 += 32) {
        // cooperative im2col fill: 1024 halves, 8 per thread
        for (int e = tid; e < 1024; e += 128) {
            int nn = e >> 5;
            int kk = e & 31;
            int k  = k0 + kk;
            float v = 0.0f;
            if (k < K) {
                int ci = k / 9;
                int rr = k - ci * 9;
                int dy = rr / 3;
                int dx = rr - dy * 3;
                int p  = j0 + nn;
                int y  = p / W;
                int x  = p - y * W;
                int yy = y + dy - 1;
                int xx = x + dx - 1;
                if (yy >= 0 && yy < H && xx >= 0 && xx < W)
                    v = (float)in[ci * HW + yy * W + xx];
            }
            ldsB[nn * 32 + kk] = (_Float16)v;
        }
        __syncthreads();

        // A frag: contiguous 32B per lane from padded f16 weights
        const _Float16* wrow = wh + (size_t)row * Kp + k0;
        v16h a = *(const v16h*)(wrow + 16 * hi);
        // B frags: contiguous 32B per lane from [n][k] LDS tile
        v16h b0 = *(const v16h*)(&ldsB[nloc * 32 + 16 * hi]);
        v16h b1 = *(const v16h*)(&ldsB[(16 + nloc) * 32 + 16 * hi]);

        acc0 = __builtin_amdgcn_wmma_f32_16x16x32_f16(false, a, false, b0,
                                                      (short)0, acc0, false, false);
        acc1 = __builtin_amdgcn_wmma_f32_16x16x32_f16(false, a, false, b1,
                                                      (short)0, acc1, false, false);
        __syncthreads();
    }

#pragma unroll
    for (int r = 0; r < 8; ++r) {
        int co = mBase + r + 8 * hi;
        float bb = bias[co];
        float v0 = acc0[r] + bb;
        float v1 = acc1[r] + bb;
        v0 = v0 > 0.0f ? v0 : 0.0f;
        v1 = v1 > 0.0f ? v1 : 0.0f;
        out[co * HW + j0 + nloc]      = (OutT)v0;
        out[co * HW + j0 + 16 + nloc] = (OutT)v1;
    }
}

// ---------------------------------------------------------------------------
// 2x2 stride-2 maxpool, f16 -> f16
// ---------------------------------------------------------------------------
__global__ void maxpool2_kernel(const _Float16* __restrict__ in,
                                _Float16* __restrict__ out,
                                int C, int H, int W) {
    int Ho = H >> 1, Wo = W >> 1;
    int total = C * Ho * Wo;
    int idx = blockIdx.x * blockDim.x + threadIdx.x;
    if (idx >= total) return;
    int c = idx / (Ho * Wo);
    int rem = idx - c * Ho * Wo;
    int y = rem / Wo;
    int x = rem - y * Wo;
    const _Float16* base = in + c * H * W + (2 * y) * W + 2 * x;
    float v0 = (float)base[0], v1 = (float)base[1];
    float v2 = (float)base[W], v3 = (float)base[W + 1];
    out[idx] = (_Float16)fmaxf(fmaxf(v0, v1), fmaxf(v2, v3));
}

// ---------------------------------------------------------------------------
// Per-channel mean over HW positions
// ---------------------------------------------------------------------------
__global__ void chmean_kernel(const float* __restrict__ feat,
                              float* __restrict__ mean, int HW) {
    __shared__ float s[256];
    int c = blockIdx.x;
    float sum = 0.0f;
    for (int i = threadIdx.x; i < HW; i += 256) sum += feat[c * HW + i];
    s[threadIdx.x] = sum;
    __syncthreads();
    for (int st = 128; st > 0; st >>= 1) {
        if (threadIdx.x < st) s[threadIdx.x] += s[threadIdx.x + st];
        __syncthreads();
    }
    if (threadIdx.x == 0) mean[c] = s[0] / (float)HW;
}

// ---------------------------------------------------------------------------
// Per-position inverse norm of centered features
// ---------------------------------------------------------------------------
__global__ void posinv_kernel(const float* __restrict__ feat,
                              const float* __restrict__ mean,
                              float* __restrict__ inv, int C, int HW) {
    int i = blockIdx.x * blockDim.x + threadIdx.x;
    if (i >= HW) return;
    float ss = 0.0f;
    for (int c = 0; c < C; ++c) {
        float d = feat[c * HW + i] - mean[c];
        ss += d * d;
    }
    inv[i] = 1.0f / (sqrtf(ss + 1e-8f) + 1e-8f);
}

// ---------------------------------------------------------------------------
// hatT[i][c] = (feat[c,i]-mean[c]) * inv[i], TRANSPOSED f16 for the WMMA GEMM
// ---------------------------------------------------------------------------
__global__ void hatT_kernel(const float* __restrict__ feat,
                            const float* __restrict__ mean,
                            const float* __restrict__ inv,
                            _Float16* __restrict__ hatT, int C, int HW) {
    int e = blockIdx.x * blockDim.x + threadIdx.x;
    if (e >= C * HW) return;
    int c = e / HW;
    int i = e - c * HW;
    hatT[(size_t)i * C + c] = (_Float16)((feat[e] - mean[c]) * inv[i]);
}

// ---------------------------------------------------------------------------
// Fused similarity-GEMM + argmax (== argmin of cosine distance).
// Inputs transposed [HW][C] so every fragment is a contiguous load.
// Block = 256 threads (8 waves) owns a 16-row i-tile; A panel (24KB) is copied
// to LDS via gfx1250 async-to-LDS when available. Each wave sweeps 32-wide
// j-pairs: one A-frag feeds 2 WMMAs.
// ---------------------------------------------------------------------------
__global__ void nn_argmax_kernel(const _Float16* __restrict__ ahatT,
                                 const _Float16* __restrict__ bhatT,
                                 int* __restrict__ nnidx, int HW, int C) {
    __shared__ alignas(32) _Float16 ldsA[16 * 768];   // [m][c], 24 KB
    __shared__ float bestv[8 * 32 * 8];
    __shared__ int   besti[8 * 32 * 8];
    const int tid  = threadIdx.x;
    const int wave = tid >> 5;
    const int lane = tid & 31;
    const int hi   = lane >> 4;
    const int nloc = lane & 15;
    const int i0   = blockIdx.x * 16;

    const _Float16* srcA = ahatT + (size_t)i0 * C;
#if defined(HAVE_ASYNC_LDS)
    for (int e = tid * 8; e < 16 * C; e += 256 * 8) {
        __builtin_amdgcn_global_load_async_to_lds_b128(
            (as1_v4i_p)(srcA + e), (as3_v4i_p)(&ldsA[e]), 0, 0);
    }
    asm volatile("s_wait_asynccnt 0x0" ::: "memory");
#else
    for (int e = tid; e < (16 * C) / 8; e += 256)
        ((uint4*)ldsA)[e] = ((const uint4*)srcA)[e];
#endif
    __syncthreads();

    float bv[8];
    int   bi[8];
#pragma unroll
    for (int r = 0; r < 8; ++r) { bv[r] = -1e30f; bi[r] = 0; }

    const _Float16* arow = &ldsA[nloc * 768];
    const int numJP = HW / 32;   // 288 pairs of 16-wide j-tiles
    for (int jp = wave; jp < numJP; jp += 8) {
        int j0 = jp * 32;
        int jn = j0 + 8 * 32;
        if (jn < HW)  // stream next tile's columns into cache
            __builtin_prefetch(&bhatT[(size_t)(jn + nloc) * C], 0, 1);

        const _Float16* bcol0 = bhatT + (size_t)(j0 + nloc) * C;
        const _Float16* bcol1 = bhatT + (size_t)(j0 + 16 + nloc) * C;
        v8f acc0 = {}, acc1 = {};
#pragma unroll 2
        for (int k0 = 0; k0 < C; k0 += 32) {
            v16h a = *(const v16h*)(arow + k0 + 16 * hi);
            v16h b0 = *(const v16h*)(bcol0 + k0 + 16 * hi);
            v16h b1 = *(const v16h*)(bcol1 + k0 + 16 * hi);
            acc0 = __builtin_amdgcn_wmma_f32_16x16x32_f16(false, a, false, b0,
                                                          (short)0, acc0, false, false);
            acc1 = __builtin_amdgcn_wmma_f32_16x16x32_f16(false, a, false, b1,
                                                          (short)0, acc1, false, false);
        }
#pragma unroll
        for (int r = 0; r < 8; ++r) {
            if (acc0[r] > bv[r]) { bv[r] = acc0[r]; bi[r] = j0 + nloc; }
            if (acc1[r] > bv[r]) { bv[r] = acc1[r]; bi[r] = j0 + 16 + nloc; }
        }
    }

#pragma unroll
    for (int r = 0; r < 8; ++r) {
        bestv[(wave * 32 + lane) * 8 + r] = bv[r];
        besti[(wave * 32 + lane) * 8 + r] = bi[r];
    }
    __syncthreads();

    // Candidate (wave,lane,r) covers i_local = r + 8*(lane>=16).
    if (tid < 16) {
        int r = tid & 7, h = tid >> 3;
        float best = -1e30f;
        int bidx = 0;
        for (int w2 = 0; w2 < 8; ++w2)
            for (int l2 = h * 16; l2 < h * 16 + 16; ++l2) {
                float v = bestv[(w2 * 32 + l2) * 8 + r];
                if (v > best) { best = v; bidx = besti[(w2 * 32 + l2) * 8 + r]; }
            }
        nnidx[i0 + tid] = bidx;
    }
}

// ---------------------------------------------------------------------------
// Gather NN columns + per-position cosine loss; accumulate sum of (1 - cos)
// ---------------------------------------------------------------------------
__global__ void loss_kernel(const float* __restrict__ xf,
                            const float* __restrict__ sf,
                            const int* __restrict__ nnidx,
                            float* __restrict__ acc, int C, int HW) {
    __shared__ float s[256];
    int i = blockIdx.x * 256 + threadIdx.x;
    float contrib = 0.0f;
    if (i < HW) {
        int j = nnidx[i];
        float dot = 0.0f, na = 0.0f, nb = 0.0f;
        for (int c = 0; c < C; ++c) {
            float a = xf[c * HW + i];
            float b = sf[c * HW + j];
            dot += a * b;
            na += a * a;
            nb += b * b;
        }
        float cs = dot / ((sqrtf(na) + 1e-8f) * (sqrtf(nb) + 1e-8f));
        contrib = 1.0f - cs;
    }
    s[threadIdx.x] = contrib;
    __syncthreads();
    for (int st = 128; st > 0; st >>= 1) {
        if (threadIdx.x < st) s[threadIdx.x] += s[threadIdx.x + st];
        __syncthreads();
    }
    if (threadIdx.x == 0) atomicAdd(acc, s[0]);
}

__global__ void zero_kernel(float* p) { p[0] = 0.0f; }
__global__ void final_kernel(const float* acc, float* out, float invN) {
    out[0] = acc[0] * invN;
}

// ---------------------------------------------------------------------------
// Host-side orchestration
// ---------------------------------------------------------------------------
extern "C" void kernel_launch(void* const* d_in, const int* in_sizes, int n_in,
                              void* d_out, int out_size, void* d_ws, size_t ws_size,
                              hipStream_t stream) {
    const float* imgs[2] = {(const float*)d_in[0], (const float*)d_in[1]};
    const float* Wt[7];
    const float* Bs[7];
    for (int i = 0; i < 7; ++i) {
        Wt[i] = (const float*)d_in[2 + 2 * i];
        Bs[i] = (const float*)d_in[3 + 2 * i];
    }

    const int HW3 = 96 * 96;   // 9216
    const int C3  = 768;
    const int Cin_[7]  = {3, 64, 64, 128, 128, 256, 256};
    const int Cout_[7] = {64, 64, 128, 128, 256, 256, 256};

    char* ws = (char*)d_ws;
    size_t off = 0;
    auto alloc = [&](size_t bytes) -> char* {
        char* p = ws + off;
        off += (bytes + 255) & ~(size_t)255;
        return p;
    };
    _Float16* buf0  = (_Float16*)alloc((size_t)64 * 384 * 384 * sizeof(_Float16));
    _Float16* buf1  = (_Float16*)alloc((size_t)64 * 384 * 384 * sizeof(_Float16));
    float*    xfeat = (float*)alloc((size_t)C3 * HW3 * sizeof(float));
    float*    sfeat = (float*)alloc((size_t)C3 * HW3 * sizeof(float));
    _Float16* wh[7];
    int       Kp[7];
    for (int i = 0; i < 7; ++i) {
        int K = Cin_[i] * 9;
        Kp[i] = (K + 31) & ~31;
        wh[i] = (_Float16*)alloc((size_t)Cout_[i] * Kp[i] * sizeof(_Float16));
    }
    float* amean = (float*)alloc(768 * sizeof(float));
    float* bmean = (float*)alloc(768 * sizeof(float));
    float* ainv  = (float*)alloc(HW3 * sizeof(float));
    float* binv  = (float*)alloc(HW3 * sizeof(float));
    int*   nnidx = (int*)alloc(HW3 * sizeof(int));
    float* accp  = (float*)alloc(256);
    // transposed hats reuse the (then free) activation ping-pong buffers
    _Float16* ahatT = buf0;
    _Float16* bhatT = buf1;

    // weight pre-conversion (f32 -> padded f16)
    for (int i = 0; i < 7; ++i) {
        int total = Cout_[i] * Kp[i];
        cvt_weights_kernel<<<(total + 255) / 256, 256, 0, stream>>>(
            Wt[i], wh[i], Cin_[i] * 9, Kp[i], total);
    }

    for (int img = 0; img < 2; ++img) {
        const float* src = imgs[img];
        float* feat = (img == 0) ? xfeat : sfeat;

        norm_input_kernel<<<(3 * 147456 + 255) / 256, 256, 0, stream>>>(src, buf0, 147456);

        conv3x3_wmma<_Float16, _Float16><<<dim3(4608, 1), 128, 0, stream>>>(
            buf0, wh[0], Bs[0], buf1, 3, 64, 384, 384, Kp[0]);
        conv3x3_wmma<_Float16, _Float16><<<dim3(4608, 1), 128, 0, stream>>>(
            buf1, wh[1], Bs[1], buf0, 64, 64, 384, 384, Kp[1]);
        maxpool2_kernel<<<(64 * 192 * 192 + 255) / 256, 256, 0, stream>>>(
            buf0, buf1, 64, 384, 384);

        conv3x3_wmma<_Float16, _Float16><<<dim3(1152, 2), 128, 0, stream>>>(
            buf1, wh[2], Bs[2], buf0, 64, 128, 192, 192, Kp[2]);
        conv3x3_wmma<_Float16, _Float16><<<dim3(1152, 2), 128, 0, stream>>>(
            buf0, wh[3], Bs[3], buf1, 128, 128, 192, 192, Kp[3]);
        maxpool2_kernel<<<(128 * 96 * 96 + 255) / 256, 256, 0, stream>>>(
            buf1, buf0, 128, 192, 192);

        conv3x3_wmma<_Float16, float><<<dim3(288, 4), 128, 0, stream>>>(
            buf0, wh[4], Bs[4], feat, 128, 256, 96, 96, Kp[4]);
        conv3x3_wmma<float, float><<<dim3(288, 4), 128, 0, stream>>>(
            feat, wh[5], Bs[5], feat + 256 * HW3, 256, 256, 96, 96, Kp[5]);
        conv3x3_wmma<float, float><<<dim3(288, 4), 128, 0, stream>>>(
            feat + 256 * HW3, wh[6], Bs[6], feat + 512 * HW3, 256, 256, 96, 96, Kp[6]);
    }

    chmean_kernel<<<768, 256, 0, stream>>>(xfeat, amean, HW3);
    chmean_kernel<<<768, 256, 0, stream>>>(sfeat, bmean, HW3);
    posinv_kernel<<<(HW3 + 255) / 256, 256, 0, stream>>>(xfeat, amean, ainv, C3, HW3);
    posinv_kernel<<<(HW3 + 255) / 256, 256, 0, stream>>>(sfeat, bmean, binv, C3, HW3);
    hatT_kernel<<<(C3 * HW3 + 255) / 256, 256, 0, stream>>>(xfeat, amean, ainv, ahatT, C3, HW3);
    hatT_kernel<<<(C3 * HW3 + 255) / 256, 256, 0, stream>>>(sfeat, bmean, binv, bhatT, C3, HW3);

    nn_argmax_kernel<<<HW3 / 16, 256, 0, stream>>>(ahatT, bhatT, nnidx, HW3, C3);

    zero_kernel<<<1, 1, 0, stream>>>(accp);
    loss_kernel<<<(HW3 + 255) / 256, 256, 0, stream>>>(xfeat, sfeat, nnidx, accp, C3, HW3);
    final_kernel<<<1, 1, 0, stream>>>(accp, (float*)d_out, 1.0f / (float)HW3);
}
